// OwnPCT_33097017983617
// MI455X (gfx1250) — matrix operand order repaired and checked by reading, not compiled, not run on hardware
//
#include <hip/hip_runtime.h>
#include <hip/hip_bf16.h>

// ---------------------------------------------------------------------------
// CDNA5 (gfx1250) PCT forward. wave32.
//  - All GEMMs: v_wmma_f32_16x16x32_f16 (f16 in, f32 accumulate)
//  - B operands staged to LDS with the Tensor Data Mover (tensor_load_to_lds),
//    double buffered, synced with s_wait_tensorcnt. TDM OOB reads return 0,
//    which handles all N/K edge padding in hardware.
//  - Each GEMM wave computes a 32x32 output tile: 2 A fragments x 2 B
//    fragments = 4 WMMAs per 32-deep k-step per TDM tile.
// ---------------------------------------------------------------------------

typedef __attribute__((ext_vector_type(16))) _Float16 v16h;
typedef __attribute__((ext_vector_type(8)))  float    v8f;
typedef __attribute__((ext_vector_type(4)))  unsigned int tdm_v4u;
typedef __attribute__((ext_vector_type(8)))  int          tdm_v8i;
typedef __attribute__((ext_vector_type(4)))  int          tdm_v4i;

#define USE_TDM 1

#define BATCH 4
#define NPTS  1620
#define NBCOL (BATCH * NPTS)      // 6480
#define CCH   128
#define KNN   32
#define EPSBN 1e-5f

__device__ inline v8f wmma_f16(v16h a, v16h b, v8f c) {
  return __builtin_amdgcn_wmma_f32_16x16x32_f16(false, a, false, b, (short)0, c,
                                                false, false);
}

// ---------------------------------------------------------------------------
// A fragment (16x32 f16). ISA layout: lanes 0-15 rows 0-15, K halves
// {0-7,16-23}; lanes 16-31 same rows, K halves {8-15,24-31}.
// ---------------------------------------------------------------------------
__device__ inline v16h load_a_fast(const float* __restrict__ A, int a_rs, int a_cs,
                                   int m0, int k0) {
  int lane = threadIdx.x & 31;
  int row  = m0 + (lane & 15);
  int koff = (lane >> 4) * 8;
  v16h a;
  if (a_cs == 1 && (a_rs & 3) == 0) {
    const float4* p0 = (const float4*)(A + (size_t)row * a_rs + k0 + koff);
    const float4* p1 = (const float4*)(A + (size_t)row * a_rs + k0 + 16 + koff);
    float4 x0 = p0[0], x1 = p0[1];
    float4 y0 = p1[0], y1 = p1[1];
    a[0] = (_Float16)x0.x; a[1] = (_Float16)x0.y; a[2] = (_Float16)x0.z; a[3] = (_Float16)x0.w;
    a[4] = (_Float16)x1.x; a[5] = (_Float16)x1.y; a[6] = (_Float16)x1.z; a[7] = (_Float16)x1.w;
    a[8]  = (_Float16)y0.x; a[9]  = (_Float16)y0.y; a[10] = (_Float16)y0.z; a[11] = (_Float16)y0.w;
    a[12] = (_Float16)y1.x; a[13] = (_Float16)y1.y; a[14] = (_Float16)y1.z; a[15] = (_Float16)y1.w;
  } else {
#pragma unroll
    for (int i = 0; i < 8; ++i) {
      a[i]     = (_Float16)A[(size_t)row * a_rs + (size_t)(k0 + koff + i) * a_cs];
      a[i + 8] = (_Float16)A[(size_t)row * a_rs + (size_t)(k0 + 16 + koff + i) * a_cs];
    }
  }
  return a;
}

// Guarded: edge tiles only.
__device__ inline v16h load_a_guard(const float* __restrict__ A, int a_rs, int a_cs,
                                    int m0, int k0, int M, int K) {
  int lane = threadIdx.x & 31;
  int row  = m0 + (lane & 15);
  int koff = (lane >> 4) * 8;
  bool mok = row < M;
  v16h a;
#pragma unroll
  for (int i = 0; i < 8; ++i) {
    int k1 = k0 + koff + i;
    int k2 = k0 + 16 + koff + i;
    a[i]     = (mok && k1 < K) ? (_Float16)A[(size_t)row * a_rs + (size_t)k1 * a_cs] : (_Float16)0.f;
    a[i + 8] = (mok && k2 < K) ? (_Float16)A[(size_t)row * a_rs + (size_t)k2 * a_cs] : (_Float16)0.f;
  }
  return a;
}

// B fragment (32x16 f16) from LDS f16 tile, leading dim ld (rows = K).
__device__ inline v16h load_b_frag_lds16(const _Float16* Bs, int k0, int n0, int ld) {
  int lane = threadIdx.x & 31;
  int col  = n0 + (lane & 15);
  int kb   = k0 + (lane >> 4) * 8;
  v16h b;
#pragma unroll
  for (int i = 0; i < 16; ++i) b[i] = Bs[(kb + i + (i >= 8 ? 8 : 0)) * ld + col];
  return b;
}

// B fragment (32x16 f16) from LDS f32 32x32 tile (TDM-staged).
__device__ inline v16h load_b_frag_lds32(const float* Bs, int nrel) {
  int lane = threadIdx.x & 31;
  int col  = nrel + (lane & 15);
  int kb   = (lane >> 4) * 8;
  v16h b;
#pragma unroll
  for (int i = 0; i < 16; ++i) b[i] = (_Float16)Bs[(kb + i + (i >= 8 ? 8 : 0)) * 32 + col];
  return b;
}

#if !USE_TDM
// Fallback: B fragment straight from global (row-major, ld = b_ld).
__device__ inline v16h load_b_guard(const float* __restrict__ Bm, int b_ld,
                                    int k0, int n0, int K, int Ntot) {
  int lane = threadIdx.x & 31;
  int col  = n0 + (lane & 15);
  int kb   = k0 + (lane >> 4) * 16;
  bool nok = col < Ntot;
  v16h b;
#pragma unroll
  for (int i = 0; i < 16; ++i) {
    int k = kb + i;
    b[i] = (nok && k < K) ? (_Float16)Bm[(size_t)k * b_ld + col] : (_Float16)0.f;
  }
  return b;
}
#endif

#if USE_TDM
// ---------------------------------------------------------------------------
// TDM: DMA a 32(rows) x 32(cols) f32 tile from global (row stride = ld elems)
// into LDS at byte offset lds_off. Dims are the *remaining* extent so the
// hardware zero-fills out-of-bounds reads.
// ---------------------------------------------------------------------------
__device__ inline void tdm_load_b_tile(const float* __restrict__ g, int rem_cols,
                                       int rem_rows, int ld, unsigned lds_off) {
  unsigned long long ga = (unsigned long long)(uintptr_t)g;
  unsigned d0 = (unsigned)rem_cols;          // tensor_dim0 (x extent)
  unsigned d1 = (unsigned)rem_rows;          // tensor_dim1 (y extent)
  unsigned long long st = (unsigned long long)(unsigned)ld;  // dim0 stride (elems)
  tdm_v4u g0;
  g0[0] = 1u;                                             // count=1, user mode
  g0[1] = lds_off;                                        // lds_addr (bytes)
  g0[2] = (unsigned)(ga & 0xffffffffull);                 // global_addr[31:0]
  g0[3] = (unsigned)((ga >> 32) & 0x1ffffffull) | (2u << 30);  // addr[56:32] | type=2
  tdm_v8i g1;
  g1[0] = (int)(2u << 16);                                // data_size=4B
  g1[1] = (int)(d0 << 16);                                // dim0[15:0] @ bits63:48
  g1[2] = (int)((d0 >> 16) | (d1 << 16));                 // dim0[31:16] | dim1[15:0]
  g1[3] = (int)((d1 >> 16) | (32u << 16));                // dim1[31:16] | tile_dim0=32
  g1[4] = (int)32u;                                       // tile_dim1=32, tile_dim2=0
  g1[5] = (int)(st & 0xffffffffull);                      // dim0_stride[31:0]
  g1[6] = (int)((st >> 32) & 0xffffull);                  // dim0_stride[47:32]
  g1[7] = 0;
  tdm_v4i z4 = {0, 0, 0, 0};
#if defined(__clang_major__) && (__clang_major__ >= 23)
  tdm_v8i z8 = {0, 0, 0, 0, 0, 0, 0, 0};
  __builtin_amdgcn_tensor_load_to_lds(g0, g1, z4, z4, z8, 0);
#else
  __builtin_amdgcn_tensor_load_to_lds(g0, g1, z4, z4, 0);
#endif
}
#endif

// ---------------------------------------------------------------------------
// Generic strided WMMA GEMM: C[m,n] = sum_k A[m,k]*B[k,n] (+ bias[m]).
// A: element (m,k) at A[m*a_rs + k*a_cs].  B: row-major [K][b_ld].
// 256 threads = 8 waves; each wave owns a 32(M) x 32(N) output tile; B tiles
// TDM-staged into per-wave LDS, double buffered; 4 WMMAs per k-step.
// ---------------------------------------------------------------------------
__global__ __launch_bounds__(256) void gemm_wmma_kernel(
    const float* __restrict__ A, int a_rs, int a_cs,
    const float* __restrict__ Bm, int b_ld,
    float* __restrict__ Cm, int c_rs,
    int M, int K, int Ntot, const float* __restrict__ bias) {
  __shared__ float s_b[8 * 2 * 32 * 32];  // 8 waves x double buffer x 4KB
  int wave = threadIdx.x >> 5;
  int lane = threadIdx.x & 31;
  int m0 = blockIdx.y * 32;
  int m1 = m0 + 16;
  int n0 = (blockIdx.x * 8 + wave) * 32;
  if (m0 >= M || n0 >= Ntot) return;  // wave-uniform exit

  float* wbuf = &s_b[wave * 2048];
  v8f acc00 = {}, acc01 = {}, acc10 = {}, acc11 = {};
  int nsteps = (K + 31) / 32;

#if USE_TDM
  unsigned lds0 = (unsigned)(uintptr_t)(void*)wbuf;
  tdm_load_b_tile(Bm + n0, Ntot - n0, K, b_ld, lds0);
#endif
  int cur = 0;
  for (int s = 0; s < nsteps; ++s) {
    int k0 = s * 32;
#if USE_TDM
    if (s + 1 < nsteps) {
      unsigned ldsn = (unsigned)(uintptr_t)(void*)(wbuf + (cur ^ 1) * 1024);
      tdm_load_b_tile(Bm + (size_t)(k0 + 32) * b_ld + n0, Ntot - n0, K - (k0 + 32),
                      b_ld, ldsn);
      __builtin_amdgcn_s_wait_tensorcnt((short)1);
    } else {
      __builtin_amdgcn_s_wait_tensorcnt((short)0);
    }
#endif
    bool kfull = (k0 + 32 <= K);
    v16h a0, a1;
    if (m0 + 16 <= M && kfull) a0 = load_a_fast(A, a_rs, a_cs, m0, k0);
    else                       a0 = load_a_guard(A, a_rs, a_cs, m0, k0, M, K);
    if (m1 + 16 <= M && kfull) a1 = load_a_fast(A, a_rs, a_cs, m1, k0);
    else                       a1 = load_a_guard(A, a_rs, a_cs, m1, k0, M, K);
#if USE_TDM
    const float* bt = wbuf + cur * 1024;
    v16h b0 = load_b_frag_lds32(bt, 0);
    acc00 = wmma_f16(a0, b0, acc00);
    acc10 = wmma_f16(a1, b0, acc10);
    v16h b1 = load_b_frag_lds32(bt, 16);
    acc01 = wmma_f16(a0, b1, acc01);
    acc11 = wmma_f16(a1, b1, acc11);
    asm volatile("s_wait_dscnt 0" ::: "memory");  // LDS reads done before overwrite
#else
    v16h b0 = load_b_guard(Bm, b_ld, k0, n0, K, Ntot);
    acc00 = wmma_f16(a0, b0, acc00);
    acc10 = wmma_f16(a1, b0, acc10);
    v16h b1 = load_b_guard(Bm, b_ld, k0, n0 + 16, K, Ntot);
    acc01 = wmma_f16(a0, b1, acc01);
    acc11 = wmma_f16(a1, b1, acc11);
#endif
    cur ^= 1;
  }

  int row8 = 8 * (lane >> 4);
  int c0 = n0 + (lane & 15);
  int c1 = n0 + 16 + (lane & 15);
#pragma unroll
  for (int r = 0; r < 8; ++r) {
    int ma = m0 + row8 + r;
    int mb = m1 + row8 + r;
    if (ma < M) {
      float bb = bias ? bias[ma] : 0.f;
      if (c0 < Ntot) Cm[(size_t)ma * c_rs + c0] = acc00[r] + bb;
      if (c1 < Ntot) Cm[(size_t)ma * c_rs + c1] = acc01[r] + bb;
    }
    if (mb < M) {
      float bb = bias ? bias[mb] : 0.f;
      if (c0 < Ntot) Cm[(size_t)mb * c_rs + c0] = acc10[r] + bb;
      if (c1 < Ntot) Cm[(size_t)mb * c_rs + c1] = acc11[r] + bb;
    }
  }
}

// ---------------------------------------------------------------------------
// conv1: t1[c, b*N+n] = sum_{k<3} W1[c,k] * x[b,k,n]
// ---------------------------------------------------------------------------
__global__ void conv1_kernel(const float* __restrict__ x, const float* __restrict__ W1,
                             float* __restrict__ t1) {
  int i = blockIdx.x * blockDim.x + threadIdx.x;
  if (i >= CCH * NBCOL) return;
  int c = i / NBCOL, col = i % NBCOL;
  int b = col / NPTS, n = col % NPTS;
  const float* xb = x + (size_t)b * 3 * NPTS;
  t1[i] = W1[c * 3 + 0] * xb[n] + W1[c * 3 + 1] * xb[NPTS + n] +
          W1[c * 3 + 2] * xb[2 * NPTS + n];
}

// ---------------------------------------------------------------------------
// BatchNorm stats / apply
// ---------------------------------------------------------------------------
__global__ void bn_stats_kernel(const float* __restrict__ src, int NB,
                                float* __restrict__ mu, float* __restrict__ rstd) {
  int c = blockIdx.x;
  __shared__ float ss[256], sq[256];
  float s = 0.f, q = 0.f;
  for (int i = threadIdx.x; i < NB; i += 256) {
    float v = src[(size_t)c * NB + i];
    s += v; q += v * v;
  }
  ss[threadIdx.x] = s; sq[threadIdx.x] = q;
  __syncthreads();
  for (int st = 128; st > 0; st >>= 1) {
    if (threadIdx.x < st) { ss[threadIdx.x] += ss[threadIdx.x + st]; sq[threadIdx.x] += sq[threadIdx.x + st]; }
    __syncthreads();
  }
  if (threadIdx.x == 0) {
    float m = ss[0] / NB;
    float var = sq[0] / NB - m * m;
    mu[c] = m;
    rstd[c] = rsqrtf(var + EPSBN);
  }
}

__global__ void bn_apply_kernel(const float* __restrict__ src, const float* __restrict__ res,
                                float* __restrict__ dst,
                                const float* __restrict__ mu, const float* __restrict__ rstd,
                                const float* __restrict__ g, const float* __restrict__ bb,
                                int C, int NB, int act) {
  int i = blockIdx.x * blockDim.x + threadIdx.x;
  if (i >= C * NB) return;
  int c = i / NB;
  float v = g[c] * (src[i] - mu[c]) * rstd[c] + bb[c];
  if (act == 1) v = fmaxf(v, 0.f);
  else if (act == 2) v = (v > 0.f) ? v : 0.2f * v;
  dst[i] = (res ? res[i] : 0.f) + v;
}

// ---------------------------------------------------------------------------
// KNN: one wave per point; 32 rounds of wave argmin (tie -> lower index)
// ---------------------------------------------------------------------------
__global__ __launch_bounds__(256) void knn_kernel(const float* __restrict__ x,
                                                  int* __restrict__ idx) {
  int wid = (blockIdx.x * blockDim.x + threadIdx.x) >> 5;
  int lane = threadIdx.x & 31;
  if (wid >= NBCOL) return;
  int b = wid / NPTS, n = wid % NPTS;
  const float* xb = x + (size_t)b * 3 * NPTS;
  float cx = xb[n], cy = xb[NPTS + n], cz = xb[2 * NPTS + n];
  const int NLOC = (NPTS + 31) / 32;  // 51
  float dl[51];
  for (int t = 0; t < NLOC; ++t) {
    int j = lane + (t << 5);
    if (j < NPTS) {
      float dx = cx - xb[j], dy = cy - xb[NPTS + j], dz = cz - xb[2 * NPTS + j];
      dl[t] = dx * dx + dy * dy + dz * dz;
    } else dl[t] = 3.0e38f;
  }
  for (int k = 0; k < KNN; ++k) {
    float bv = 3.0e38f; int bt = 0;
    for (int t = 0; t < NLOC; ++t)
      if (dl[t] < bv) { bv = dl[t]; bt = t; }
    int bj = (bt << 5) + lane;
    for (int off = 16; off > 0; off >>= 1) {
      float ov = __shfl_xor(bv, off, 32);
      int   oj = __shfl_xor(bj, off, 32);
      if (ov < bv || (ov == bv && oj < bj)) { bv = ov; bj = oj; }
    }
    if ((bj & 31) == lane) dl[bj >> 5] = 3.0e38f;
    if (lane == 0) idx[(size_t)wid * KNN + k] = bj;
  }
}

// ---------------------------------------------------------------------------
// Neighbor MLP: one block (8 waves) per point. A fragments reused across both
// 16-col subtiles (2 WMMAs per A load).
// ---------------------------------------------------------------------------
__global__ __launch_bounds__(256) void neigh_mlp_kernel(
    const float* __restrict__ f, const int* __restrict__ idx,
    const float* __restrict__ Wg1, const float* __restrict__ Wg2,
    const float* __restrict__ mu1, const float* __restrict__ rs1,
    const float* __restrict__ gg1, const float* __restrict__ bg1,
    const float* __restrict__ mu2, const float* __restrict__ rs2,
    const float* __restrict__ gg2, const float* __restrict__ bg2,
    float* __restrict__ stats1, float* __restrict__ stats2,
    float* __restrict__ f0, int phase) {
  __shared__ _Float16 s_nf[256 * KNN];
  __shared__ _Float16 s_y1[256 * KNN];
  __shared__ float    s_v[128 * KNN];
  __shared__ float    s_sum[256], s_sq[256];
  __shared__ int      s_idx[KNN];

  int p = blockIdx.x;               // column index (= b*N + n)
  int b = p / NPTS;
  int tid = threadIdx.x;
  if (tid < KNN) s_idx[tid] = b * NPTS + idx[(size_t)p * KNN + tid];
  s_sum[tid] = 0.f; s_sq[tid] = 0.f;
  __syncthreads();

  // build nf (256 x 32) in LDS as f16
  for (int e = tid; e < 256 * KNN; e += 256) {
    int c = e >> 5, k = e & 31;
    float v;
    if (c < 128) v = f[(size_t)c * NBCOL + s_idx[k]] - f[(size_t)c * NBCOL + p];
    else         v = f[(size_t)(c - 128) * NBCOL + p];
    s_nf[c * KNN + k] = (_Float16)v;
  }
  __syncthreads();

  int wave = tid >> 5, lane = tid & 31;
  int row8 = 8 * (lane >> 4);
  int cL = lane & 15;

  // GEMM1: u = Wg1(256x256) x nf(256x32); wave handles m-tiles {2w, 2w+1}
  for (int mi = 0; mi < 2; ++mi) {
    int m0 = (wave * 2 + mi) * 16;
    v8f a0 = {}, a1 = {};
    for (int k0 = 0; k0 < 256; k0 += 32) {
      v16h a = load_a_fast(Wg1, 256, 1, m0, k0);
      a0 = wmma_f16(a, load_b_frag_lds16(s_nf, k0, 0, KNN), a0);
      a1 = wmma_f16(a, load_b_frag_lds16(s_nf, k0, 16, KNN), a1);
    }
#pragma unroll
    for (int r = 0; r < 8; ++r) {
      int m = m0 + row8 + r;
      float u0 = a0[r], u1 = a1[r];
      if (phase == 0) {
        atomicAdd(&s_sum[m], u0 + u1);
        atomicAdd(&s_sq[m], u0 * u0 + u1 * u1);
      } else {
        float y0 = gg1[m] * (u0 - mu1[m]) * rs1[m] + bg1[m];
        float y1 = gg1[m] * (u1 - mu1[m]) * rs1[m] + bg1[m];
        s_y1[m * KNN + cL]      = (_Float16)fmaxf(y0, 0.f);
        s_y1[m * KNN + 16 + cL] = (_Float16)fmaxf(y1, 0.f);
      }
    }
  }
  __syncthreads();
  if (phase == 0) {
    if (tid < 256) {
      atomicAdd(&stats1[tid], s_sum[tid]);
      atomicAdd(&stats1[256 + tid], s_sq[tid]);
    }
    return;
  }

  // GEMM2: v = Wg2(128x256) x y1(256x32); wave handles m-tile = wave
  {
    int m0 = wave * 16;
    v8f a0 = {}, a1 = {};
    for (int k0 = 0; k0 < 256; k0 += 32) {
      v16h a = load_a_fast(Wg2, 256, 1, m0, k0);
      a0 = wmma_f16(a, load_b_frag_lds16(s_y1, k0, 0, KNN), a0);
      a1 = wmma_f16(a, load_b_frag_lds16(s_y1, k0, 16, KNN), a1);
    }
#pragma unroll
    for (int r = 0; r < 8; ++r) {
      int m = m0 + row8 + r;
      float v0 = a0[r], v1 = a1[r];
      if (phase == 1) {
        atomicAdd(&s_sum[m], v0 + v1);
        atomicAdd(&s_sq[m], v0 * v0 + v1 * v1);
      } else {
        float y0 = gg2[m] * (v0 - mu2[m]) * rs2[m] + bg2[m];
        float y1 = gg2[m] * (v1 - mu2[m]) * rs2[m] + bg2[m];
        s_v[m * KNN + cL]      = fmaxf(y0, 0.f);
        s_v[m * KNN + 16 + cL] = fmaxf(y1, 0.f);
      }
    }
  }
  __syncthreads();
  if (phase == 1) {
    if (tid < 128) {
      atomicAdd(&stats2[tid], s_sum[tid]);
      atomicAdd(&stats2[128 + tid], s_sq[tid]);
    }
    return;
  }

  // phase 2: max over K neighbors
  if (tid < 128) {
    float mx = s_v[tid * KNN];
    for (int k = 1; k < KNN; ++k) mx = fmaxf(mx, s_v[tid * KNN + k]);
    f0[(size_t)tid * NBCOL + p] = mx;
  }
}

__global__ void zero_kernel(float* __restrict__ p, int n) {
  int i = blockIdx.x * blockDim.x + threadIdx.x;
  if (i < n) p[i] = 0.f;
}

__global__ void finalize_stats_kernel(const float* __restrict__ raw, int C, float cnt,
                                      float* __restrict__ mu, float* __restrict__ rstd) {
  int c = blockIdx.x * blockDim.x + threadIdx.x;
  if (c >= C) return;
  float m = raw[c] / cnt;
  float var = raw[C + c] / cnt - m * m;
  mu[c] = m;
  rstd[c] = rsqrtf(var + EPSBN);
}

// ---------------------------------------------------------------------------
// Attention helpers
// ---------------------------------------------------------------------------
__global__ void softmax_row_kernel(float* __restrict__ att) {
  int r = blockIdx.x;
  float* row = att + (size_t)r * NPTS;
  __shared__ float red[256];
  int tid = threadIdx.x;
  float mx = -3.0e38f;
  for (int i = tid; i < NPTS; i += 256) mx = fmaxf(mx, row[i]);
  red[tid] = mx; __syncthreads();
  for (int s = 128; s > 0; s >>= 1) { if (tid < s) red[tid] = fmaxf(red[tid], red[tid + s]); __syncthreads(); }
  mx = red[0]; __syncthreads();
  float sm = 0.f;
  for (int i = tid; i < NPTS; i += 256) { float e = expf(row[i] - mx); row[i] = e; sm += e; }
  red[tid] = sm; __syncthreads();
  for (int s = 128; s > 0; s >>= 1) { if (tid < s) red[tid] += red[tid + s]; __syncthreads(); }
  float inv = 1.f / red[0];
  for (int i = tid; i < NPTS; i += 256) row[i] *= inv;
}

__global__ void colsum_kernel(const float* __restrict__ att, float* __restrict__ cs) {
  int m = blockIdx.x * blockDim.x + threadIdx.x;
  if (m >= NPTS) return;
  float s = 0.f;
  for (int n2 = 0; n2 < NPTS; ++n2) s += att[(size_t)n2 * NPTS + m];
  cs[m] = s;
}

__global__ void colscale_kernel(float* __restrict__ att, const float* __restrict__ cs) {
  int i = blockIdx.x * blockDim.x + threadIdx.x;
  if (i >= NPTS * NPTS) return;
  att[i] /= (1e-9f + cs[i % NPTS]);
}

__global__ void sub_kernel(const float* __restrict__ a, const float* __restrict__ b,
                           float* __restrict__ c, int n) {
  int i = blockIdx.x * blockDim.x + threadIdx.x;
  if (i < n) c[i] = a[i] - b[i];
}

// ---------------------------------------------------------------------------
// Final: y = leaky(bn(ylin)) transposed to (B, 640, N); then per-row max/mean
// ---------------------------------------------------------------------------
__global__ void final_y_kernel(const float* __restrict__ ylin,
                               const float* __restrict__ mu, const float* __restrict__ rstd,
                               const float* __restrict__ gL, const float* __restrict__ bL,
                               float* __restrict__ yout, int CL) {
  int i = blockIdx.x * blockDim.x + threadIdx.x;
  if (i >= CL * NBCOL) return;
  int c = i / NBCOL, col = i % NBCOL;
  int b = col / NPTS, n = col % NPTS;
  float v = gL[c] * (ylin[i] - mu[c]) * rstd[c] + bL[c];
  v = (v > 0.f) ? v : 0.2f * v;
  yout[((size_t)b * CL + c) * NPTS + n] = v;
}

__global__ void rowstat_kernel(const float* __restrict__ y, float* __restrict__ ymax,
                               float* __restrict__ ymean) {
  const float* row = y + (size_t)blockIdx.x * NPTS;
  __shared__ float rmx[256], rsm[256];
  int tid = threadIdx.x;
  float mx = -3.0e38f, sm = 0.f;
  for (int i = tid; i < NPTS; i += 256) { float v = row[i]; mx = fmaxf(mx, v); sm += v; }
  rmx[tid] = mx; rsm[tid] = sm;
  __syncthreads();
  for (int s = 128; s > 0; s >>= 1) {
    if (tid < s) { rmx[tid] = fmaxf(rmx[tid], rmx[tid + s]); rsm[tid] += rsm[tid + s]; }
    __syncthreads();
  }
  if (tid == 0) { ymax[blockIdx.x] = rmx[0]; ymean[blockIdx.x] = rsm[0] / NPTS; }
}

// ---------------------------------------------------------------------------
// Host orchestration
// ---------------------------------------------------------------------------
static inline void launch_gemm(const float* A, int a_rs, int a_cs,
                               const float* B, int b_ld,
                               float* C, int c_rs, int M, int K, int NT,
                               const float* bias, hipStream_t s) {
  dim3 grid((NT + 255) / 256, (M + 31) / 32);
  gemm_wmma_kernel<<<grid, 256, 0, s>>>(A, a_rs, a_cs, B, b_ld, C, c_rs, M, K, NT, bias);
}

extern "C" void kernel_launch(void* const* d_in, const int* in_sizes, int n_in,
                              void* d_out, int out_size, void* d_ws, size_t ws_size,
                              hipStream_t stream) {
  const float* x    = (const float*)d_in[0];
  const float* W1   = (const float*)d_in[1];
  const float* g1   = (const float*)d_in[2];
  const float* b1   = (const float*)d_in[3];
  const float* W2   = (const float*)d_in[4];
  const float* g2   = (const float*)d_in[5];
  const float* b2   = (const float*)d_in[6];
  const float* Wg1  = (const float*)d_in[7];
  const float* gg1  = (const float*)d_in[8];
  const float* bg1  = (const float*)d_in[9];
  const float* Wg2  = (const float*)d_in[10];
  const float* gg2  = (const float*)d_in[11];
  const float* bg2  = (const float*)d_in[12];
  const float* Wqk  = (const float*)d_in[13];
  const float* Wv   = (const float*)d_in[14];
  const float* bv   = (const float*)d_in[15];
  const float* Wt   = (const float*)d_in[16];
  const float* bt   = (const float*)d_in[17];
  const float* oag  = (const float*)d_in[18];
  const float* oab  = (const float*)d_in[19];
  const float* WL   = (const float*)d_in[20];
  const float* gL   = (const float*)d_in[21];
  const float* bL   = (const float*)d_in[22];

  const int NB = NBCOL;
  const size_t SEG = (size_t)CCH * NB;  // 829440

  float* w = (float*)d_ws;
  size_t o = 0;
  float* buf_h    = w + o; o += SEG;
  float* buf_f    = w + o; o += SEG;
  float* buf_t    = w + o; o += SEG;
  float* buf_cat  = w + o; o += 5 * SEG;      // seg0=f0, seg1=out4, seg2..4=out1..3
  float* buf_qk   = w + o; o += (size_t)32 * NB;
  float* buf_v    = w + o; o += SEG;
  float* buf_att  = w + o; o += (size_t)NPTS * NPTS;
  float* buf_xr   = w + o; o += SEG;
  float* buf_diff = w + o; o += SEG;
  float* buf_ylin = w + o; o += (size_t)640 * NB;
  float* colsum   = w + o; o += NPTS;
  float* st_mu    = w + o; o += 640;
  float* st_rs    = w + o; o += 640;
  float* nm1      = w + o; o += 512;  // sum[256], sq[256]
  float* nm2      = w + o; o += 256;  // sum[128], sq[128]
  float* mu1      = w + o; o += 256;
  float* rs1      = w + o; o += 256;
  float* mu2      = w + o; o += 128;
  float* rs2      = w + o; o += 128;
  int*   idx      = (int*)(w + o); o += (size_t)NB * KNN;

  float* yout  = (float*)d_out;                       // (B,640,N)
  float* ymax  = yout + (size_t)BATCH * 640 * NPTS;   // (B,640)
  float* ymean = ymax + (size_t)BATCH * 640;          // (B,640)

  const int EL = CCH * NB;
  const int TPB = 256;

  // ---- stage 1: h = relu(bn(W1 @ x)) ----
  conv1_kernel<<<(EL + TPB - 1) / TPB, TPB, 0, stream>>>(x, W1, buf_t);
  bn_stats_kernel<<<CCH, TPB, 0, stream>>>(buf_t, NB, st_mu, st_rs);
  bn_apply_kernel<<<(EL + TPB - 1) / TPB, TPB, 0, stream>>>(buf_t, nullptr, buf_h,
      st_mu, st_rs, g1, b1, CCH, NB, 1);

  // ---- stage 2: f = relu(bn(W2 @ h)) ----
  launch_gemm(W2, CCH, 1, buf_h, NB, buf_t, NB, CCH, CCH, NB, nullptr, stream);
  bn_stats_kernel<<<CCH, TPB, 0, stream>>>(buf_t, NB, st_mu, st_rs);
  bn_apply_kernel<<<(EL + TPB - 1) / TPB, TPB, 0, stream>>>(buf_t, nullptr, buf_f,
      st_mu, st_rs, g2, b2, CCH, NB, 1);

  // ---- stage 3: KNN ----
  knn_kernel<<<(NB * 32 + TPB - 1) / TPB, TPB, 0, stream>>>(x, idx);

  // ---- stage 4: neighbor MLP (3 phases, recompute instead of 200MB scratch) ----
  zero_kernel<<<3, TPB, 0, stream>>>(nm1, 768);
  neigh_mlp_kernel<<<NB, TPB, 0, stream>>>(buf_f, idx, Wg1, Wg2,
      nullptr, nullptr, nullptr, nullptr, nullptr, nullptr, nullptr, nullptr,
      nm1, nm2, nullptr, 0);
  finalize_stats_kernel<<<1, 256, 0, stream>>>(nm1, 256, (float)NB * KNN, mu1, rs1);
  neigh_mlp_kernel<<<NB, TPB, 0, stream>>>(buf_f, idx, Wg1, Wg2,
      mu1, rs1, gg1, bg1, nullptr, nullptr, nullptr, nullptr,
      nm1, nm2, nullptr, 1);
  finalize_stats_kernel<<<1, 128, 0, stream>>>(nm2, 128, (float)NB * KNN, mu2, rs2);
  neigh_mlp_kernel<<<NB, TPB, 0, stream>>>(buf_f, idx, Wg1, Wg2,
      mu1, rs1, gg1, bg1, mu2, rs2, gg2, bg2,
      nm1, nm2, buf_cat /*seg0 = f0*/, 2);

  // ---- stage 5: 4 offset-attention layers ----
  float* seg_in[4]  = { buf_cat + 0 * SEG, buf_cat + 2 * SEG, buf_cat + 3 * SEG, buf_cat + 4 * SEG };
  float* seg_out[4] = { buf_cat + 2 * SEG, buf_cat + 3 * SEG, buf_cat + 4 * SEG, buf_cat + 1 * SEG };

  for (int L = 0; L < 4; ++L) {
    const float* hcur = seg_in[L];
    float* hout = seg_out[L];
    const float* Wqk_i = Wqk + (size_t)L * 32 * CCH;
    const float* Wv_i  = Wv  + (size_t)L * CCH * CCH;
    const float* bv_i  = bv  + (size_t)L * CCH;
    const float* Wt_i  = Wt  + (size_t)L * CCH * CCH;
    const float* bt_i  = bt  + (size_t)L * CCH;
    const float* g_i   = oag + (size_t)L * CCH;
    const float* b_i   = oab + (size_t)L * CCH;

    // qk = Wqk @ h (32 x NB);  xv = Wv @ h + bv (128 x NB)
    launch_gemm(Wqk_i, CCH, 1, hcur, NB, buf_qk, NB, 32, CCH, NB, nullptr, stream);
    launch_gemm(Wv_i, CCH, 1, hcur, NB, buf_v, NB, CCH, CCH, NB, bv_i, stream);

    for (int b = 0; b < BATCH; ++b) {
      const float* qkb = buf_qk + (size_t)b * NPTS;
      // energy = qk_b^T (N x 32) @ qk_b (32 x N)   (A transposed via strides)
      launch_gemm(qkb, 1, NB, qkb, NB, buf_att, NPTS, NPTS, 32, NPTS, nullptr, stream);
      softmax_row_kernel<<<NPTS, TPB, 0, stream>>>(buf_att);
      colsum_kernel<<<(NPTS + TPB - 1) / TPB, TPB, 0, stream>>>(buf_att, colsum);
      colscale_kernel<<<(NPTS * NPTS + TPB - 1) / TPB, TPB, 0, stream>>>(buf_att, colsum);
      // xr = xv_b (128 x N) @ attn (N x N)
      launch_gemm(buf_v + (size_t)b * NPTS, NB, 1, buf_att, NPTS,
                  buf_xr + (size_t)b * NPTS, NB, CCH, NPTS, NPTS, nullptr, stream);
    }

    // t = Wt @ (h - xr) + bt ; hout = h + relu(bn(t))
    sub_kernel<<<(EL + TPB - 1) / TPB, TPB, 0, stream>>>(hcur, buf_xr, buf_diff, EL);
    launch_gemm(Wt_i, CCH, 1, buf_diff, NB, buf_t, NB, CCH, CCH, NB, bt_i, stream);
    bn_stats_kernel<<<CCH, TPB, 0, stream>>>(buf_t, NB, st_mu, st_rs);
    bn_apply_kernel<<<(EL + TPB - 1) / TPB, TPB, 0, stream>>>(buf_t, hcur, hout,
        st_mu, st_rs, g_i, b_i, CCH, NB, 1);
  }

  // ---- stage 6: y = leaky(bn(WL @ cat)) ----
  launch_gemm(WL, 640, 1, buf_cat, NB, buf_ylin, NB, 640, 640, NB, nullptr, stream);
  bn_stats_kernel<<<640, TPB, 0, stream>>>(buf_ylin, NB, st_mu, st_rs);
  final_y_kernel<<<(640 * NB + TPB - 1) / TPB, TPB, 0, stream>>>(buf_ylin, st_mu, st_rs,
      gL, bL, yout, 640);
  rowstat_kernel<<<BATCH * 640, TPB, 0, stream>>>(yout, ymax, ymean);

  (void)in_sizes; (void)n_in; (void)out_size; (void)ws_size;
}
